// TemporalDeformableAttention_37374805410517
// MI455X (gfx1250) — compile-verified, matrix operand order checked by reading, not run on hardware
//
#include <hip/hip_runtime.h>

// ---------------------------------------------------------------------------
// TemporalDeformableAttention block for MI455X (gfx1250, wave32, WMMA + TDM).
// All GEMMs + both attention matmuls run on v_wmma_f32_16x16x32_bf16
// (bf16 operands, fp32 accumulate). GEMM tiles staged into LDS via the
// Tensor Data Mover (tensor_load_to_lds / s_wait_tensorcnt) when available.
// ---------------------------------------------------------------------------

typedef __attribute__((ext_vector_type(16))) __bf16 v16bf;
typedef __attribute__((ext_vector_type(8)))  float  v8f;

// Module constants
#define NQ   4096
#define NV   16384
#define CDIM 256
#define HEADS 8
#define DH   32
#define LVLS 4
#define PTS  4
#define FFD  1024
#define HBEV 32
#define WBEV 128

#if defined(__has_builtin)
#if __has_builtin(__builtin_amdgcn_tensor_load_to_lds) && \
    __has_builtin(__builtin_amdgcn_s_wait_tensorcnt)
#define HAVE_TDM 1
#endif
#endif

__device__ __forceinline__ unsigned short f2bf(float f) {
    union { float f; unsigned u; } x; x.f = f;
    unsigned r = x.u + 0x7FFFu + ((x.u >> 16) & 1u);
    return (unsigned short)(r >> 16);
}

// Load a 16x32 bf16 fragment from row-major memory (ld in elements).
// Layout per CDNA5 ISA 7.12.2: lane<16 -> row=lane, K = 0..7 & 16..23;
// lane>=16 -> row=lane-16, K = 8..15 & 24..31. Two contiguous 16B loads.
__device__ __forceinline__ v16bf frag_rowmajor(const unsigned short* base, int ld) {
    int lane = threadIdx.x & 31;
    int row  = lane & 15;
    int koff = (lane >> 4) << 3;            // 0 or 8
    const unsigned short* p = base + (size_t)row * ld + koff;
    union { v16bf v; uint4 q[2]; } u;
    u.q[0] = *(const uint4*)(p);
    u.q[1] = *(const uint4*)(p + 16);
    return u.v;
}

__device__ __forceinline__ v8f wmma_bf16(v16bf a, v16bf b, v8f c) {
    return __builtin_amdgcn_wmma_f32_16x16x32_bf16(
        false, a, false, b, (short)0, c, false, false);
}

#ifdef HAVE_TDM
typedef unsigned u32x4 __attribute__((ext_vector_type(4)));
typedef int      i32x4 __attribute__((ext_vector_type(4)));
typedef int      i32x8 __attribute__((ext_vector_type(8)));

// Issue a TDM 2D tile load: tile_h rows x tile_w elems (bf16, data_size=2B)
// from row-major global tensor (row stride = stride_e elements) into LDS at
// byte offset lds_off, rows stored compacted (tile_w elems each).
// Descriptor packing per CDNA5 ISA 8.3/8.4 (D# groups 0/1).
__device__ __forceinline__ void tdm_load_2d(unsigned lds_off, const void* gptr,
                                            unsigned tensor_w, unsigned tensor_h,
                                            unsigned tile_w, unsigned tile_h,
                                            unsigned stride_e) {
    unsigned long long ga = (unsigned long long)gptr;
    u32x4 g0;
    g0[0] = 1u;                                        // count=1, user mode
    g0[1] = lds_off;                                   // lds_addr (bytes)
    g0[2] = (unsigned)(ga & 0xFFFFFFFFu);              // global_addr[31:0]
    g0[3] = (unsigned)((ga >> 32) & 0x01FFFFFFu)       // global_addr[56:32]
          | (2u << 30);                                // type = 2 ("image")
    i32x8 g1;
    g1[0] = (int)(1u << 16);                           // data_size=1 (2 bytes)
    g1[1] = (int)((tensor_w & 0xFFFFu) << 16);         // tensor_dim0[15:0]
    g1[2] = (int)((tensor_w >> 16) |
                  ((tensor_h & 0xFFFFu) << 16));       // dim0[31:16]|dim1[15:0]
    g1[3] = (int)((tensor_h >> 16) | (tile_w << 16));  // dim1[31:16]|tile_dim0
    g1[4] = (int)tile_h;                               // tile_dim1 (tile_dim2=0)
    g1[5] = (int)stride_e;                             // tensor_dim0_stride[31:0]
    g1[6] = 0;                                         // stride0[47:32]|stride1 lo
    g1[7] = 0;
    i32x4 z4 = {0, 0, 0, 0};
#if __clang_major__ >= 23
    i32x8 z8 = {0, 0, 0, 0, 0, 0, 0, 0};
    __builtin_amdgcn_tensor_load_to_lds(g0, g1, z4, z4, z8, 0);
#else
    __builtin_amdgcn_tensor_load_to_lds(g0, g1, z4, z4, 0);
#endif
}
#endif  // HAVE_TDM

// ---------------------------------------------------------------------------
// Weight transpose + convert: wT[n*K + k] = bf16(w[k*N + n])
// ---------------------------------------------------------------------------
__global__ void wtr_kernel(const float* __restrict__ w, unsigned short* __restrict__ wT,
                           int K, int N) {
    int idx = blockIdx.x * 256 + threadIdx.x;
    if (idx >= K * N) return;
    int n = idx / K, k = idx - n * K;
    wT[idx] = f2bf(w[(size_t)k * N + n]);
}

// fp32 -> bf16 flat convert
__global__ void cvt_kernel(const float* __restrict__ in, unsigned short* __restrict__ out,
                           int n) {
    int idx = blockIdx.x * 256 + threadIdx.x;
    if (idx < n) out[idx] = f2bf(in[idx]);
}

// ---------------------------------------------------------------------------
// LayerNorm over last dim (=256) -> bf16 output. One wave per row.
// ---------------------------------------------------------------------------
__global__ __launch_bounds__(256)
void ln_bf16_kernel(const float* __restrict__ x, const float* __restrict__ g,
                    const float* __restrict__ b, unsigned short* __restrict__ out, int M) {
    int wave = threadIdx.x >> 5, lane = threadIdx.x & 31;
    int row = blockIdx.x * 8 + wave;
    if (row >= M) return;
    const float* xr = x + (size_t)row * CDIM;
    float v[8]; float s = 0.f;
    #pragma unroll
    for (int i = 0; i < 8; ++i) { v[i] = xr[lane + 32 * i]; s += v[i]; }
    #pragma unroll
    for (int m = 16; m >= 1; m >>= 1) s += __shfl_xor(s, m, 32);
    float mean = s * (1.f / 256.f);
    float q = 0.f;
    #pragma unroll
    for (int i = 0; i < 8; ++i) { float d = v[i] - mean; q += d * d; }
    #pragma unroll
    for (int m = 16; m >= 1; m >>= 1) q += __shfl_xor(q, m, 32);
    float rstd = rsqrtf(q * (1.f / 256.f) + 1e-5f);
    #pragma unroll
    for (int i = 0; i < 8; ++i) {
        int d = lane + 32 * i;
        out[(size_t)row * CDIM + d] = f2bf((v[i] - mean) * rstd * g[d] + b[d]);
    }
}

// ---------------------------------------------------------------------------
// Generic WMMA GEMM: C[M,N] = A[M,K](bf16) @ Bt[N,K](bf16)^T + bias + resid
// Block 256 threads (8 waves); tile 128(M) x 64(N); K-step 32.
// Tiles staged to LDS via TDM when available, else via b128 vector copies.
// ---------------------------------------------------------------------------
__global__ __launch_bounds__(256)
void gemm_bf16_kernel(const unsigned short* __restrict__ A,
                      const unsigned short* __restrict__ Bt,
                      const float* __restrict__ bias,
                      const float* __restrict__ resid,
                      float* __restrict__ C, int M, int N, int K) {
    __shared__ __align__(16) unsigned short sA[128 * 32];
    __shared__ __align__(16) unsigned short sB[64 * 32];
    int wave = threadIdx.x >> 5, lane = threadIdx.x & 31;
    int bm = blockIdx.x * 128;
    int bn = blockIdx.y * 64;

    v8f acc[4] = {v8f{}, v8f{}, v8f{}, v8f{}};
    for (int k0 = 0; k0 < K; k0 += 32) {
#ifdef HAVE_TDM
        if (wave == 0) {
            // Tensor Data Mover: DMA both tiles straight into LDS.
            tdm_load_2d((unsigned)(size_t)(&sA[0]),
                        A + (size_t)bm * K + k0, (unsigned)K, (unsigned)(M - bm),
                        32u, 128u, (unsigned)K);
            tdm_load_2d((unsigned)(size_t)(&sB[0]),
                        Bt + (size_t)bn * K + k0, (unsigned)K, (unsigned)(N - bn),
                        32u, 64u, (unsigned)K);
            __builtin_amdgcn_s_wait_tensorcnt((short)0);
        }
        __syncthreads();
#else
        {   // A tile: 128 rows x 32 cols; each thread copies 16 elems (32B)
            int t = threadIdx.x;
            int row = t >> 1, col = (t & 1) * 16;
            const uint4* src = (const uint4*)(A + (size_t)(bm + row) * K + k0 + col);
            uint4* dst = (uint4*)(sA + row * 32 + col);
            dst[0] = src[0]; dst[1] = src[1];
        }
        if (threadIdx.x < 128) {  // B tile: 64 rows x 32 cols
            int t = threadIdx.x;
            int row = t >> 1, col = (t & 1) * 16;
            const uint4* src = (const uint4*)(Bt + (size_t)(bn + row) * K + k0 + col);
            uint4* dst = (uint4*)(sB + row * 32 + col);
            dst[0] = src[0]; dst[1] = src[1];
        }
        __syncthreads();
#endif
        v16bf af = frag_rowmajor(sA + wave * 16 * 32, 32);
        #pragma unroll
        for (int nt = 0; nt < 4; ++nt) {
            v16bf bf = frag_rowmajor(sB + nt * 16 * 32, 32);
            acc[nt] = wmma_bf16(af, bf, acc[nt]);
        }
        __syncthreads();
    }
    int half = lane >> 4, col = lane & 15;
    #pragma unroll
    for (int nt = 0; nt < 4; ++nt) {
        int n = bn + nt * 16 + col;
        float bv = bias ? bias[n] : 0.f;
        #pragma unroll
        for (int r = 0; r < 8; ++r) {
            int m = bm + wave * 16 + half * 8 + r;
            size_t idx = (size_t)m * N + n;
            float v = acc[nt][r] + bv;
            if (resid) v += resid[idx];
            C[idx] = v;
        }
    }
}

// ---------------------------------------------------------------------------
// Split qkv [4096][768] fp32 into per-head bf16: Qh/Kh [h][n][32], Vt [h][32][n]
// ---------------------------------------------------------------------------
__global__ __launch_bounds__(256)
void qkv_split_kernel(const float* __restrict__ qkv, unsigned short* __restrict__ Qh,
                      unsigned short* __restrict__ Kh, unsigned short* __restrict__ Vt) {
    int idx = blockIdx.x * 256 + threadIdx.x;     // over NQ*CDIM
    int n = idx >> 8, c = idx & 255;
    int h = c >> 5, d = c & 31;
    float q = qkv[(size_t)n * 768 + c];
    float k = qkv[(size_t)n * 768 + 256 + c];
    float v = qkv[(size_t)n * 768 + 512 + c];
    Qh[((size_t)h * NQ + n) * DH + d] = f2bf(q);
    Kh[((size_t)h * NQ + n) * DH + d] = f2bf(k);
    Vt[(size_t)h * DH * NQ + (size_t)d * NQ + n] = f2bf(v);
}

// ---------------------------------------------------------------------------
// Flash self-attention, d=32. Grid (NQ/128, HEADS), 256 threads (8 waves),
// each wave owns 16 queries. S and P*V both on WMMA bf16.
// ---------------------------------------------------------------------------
__global__ __launch_bounds__(256)
void attn_kernel(const unsigned short* __restrict__ Qh,
                 const unsigned short* __restrict__ Kh,
                 const unsigned short* __restrict__ Vt,
                 float* __restrict__ xsa) {              // [NQ][256]
    __shared__ __align__(16) unsigned short sP[8][16 * 32];
    int wave = threadIdx.x >> 5, lane = threadIdx.x & 31;
    int h = blockIdx.y;
    int qbase = blockIdx.x * 128 + wave * 16;
    const unsigned short* Q = Qh + ((size_t)h * NQ + qbase) * DH;
    const unsigned short* K = Kh + (size_t)h * NQ * DH;
    const unsigned short* V = Vt + (size_t)h * DH * NQ;

    v16bf qf = frag_rowmajor(Q, DH);
    v8f o0 = {}, o1 = {};
    float m[8], l[8];
    #pragma unroll
    for (int r = 0; r < 8; ++r) { m[r] = -1e30f; l[r] = 0.f; }
    const float scale = 0.17677669529663687f;  // 1/sqrt(32)
    int half = lane >> 4, col = lane & 15;
    unsigned short* sp = sP[wave];

    for (int kb = 0; kb < NQ; kb += 32) {
        v16bf kf0 = frag_rowmajor(K + (size_t)kb * DH, DH);
        v16bf kf1 = frag_rowmajor(K + (size_t)(kb + 16) * DH, DH);
        v8f s0 = {}, s1 = {};
        s0 = wmma_bf16(qf, kf0, s0);
        s1 = wmma_bf16(qf, kf1, s1);
        #pragma unroll
        for (int r = 0; r < 8; ++r) {
            float a0 = s0[r] * scale, a1 = s1[r] * scale;
            float mx = fmaxf(a0, a1);
            #pragma unroll
            for (int msk = 8; msk >= 1; msk >>= 1) mx = fmaxf(mx, __shfl_xor(mx, msk, 32));
            float mn = fmaxf(m[r], mx);
            float corr = __expf(m[r] - mn);
            m[r] = mn;
            float p0 = __expf(a0 - mn);
            float p1 = __expf(a1 - mn);
            float rs = p0 + p1;
            #pragma unroll
            for (int msk = 8; msk >= 1; msk >>= 1) rs += __shfl_xor(rs, msk, 32);
            l[r] = l[r] * corr + rs;
            o0[r] *= corr; o1[r] *= corr;
            // C-layout (row = r + 8*half, col = lane&15) -> row-major P in LDS
            sp[(r + 8 * half) * 32 + col]      = f2bf(p0);
            sp[(r + 8 * half) * 32 + 16 + col] = f2bf(p1);
        }
        asm volatile("s_wait_dscnt 0" ::: "memory");
        v16bf pf  = frag_rowmajor(sp, 32);
        v16bf vf0 = frag_rowmajor(V + kb, NQ);
        v16bf vf1 = frag_rowmajor(V + (size_t)16 * NQ + kb, NQ);
        o0 = wmma_bf16(pf, vf0, o0);
        o1 = wmma_bf16(pf, vf1, o1);
    }
    #pragma unroll
    for (int r = 0; r < 8; ++r) {
        int q = qbase + r + 8 * half;
        float inv = 1.f / l[r];
        xsa[(size_t)q * CDIM + h * DH + col]      = o0[r] * inv;
        xsa[(size_t)q * CDIM + h * DH + 16 + col] = o1[r] * inv;
    }
}

// ---------------------------------------------------------------------------
// Multi-scale deformable sampling. One wave per (query, head); lane = channel.
// ---------------------------------------------------------------------------
__device__ __forceinline__ float msda_tap(const float* vl, int x, int y, float w) {
    if (x < 0 || x >= WBEV || y < 0 || y >= HBEV) return 0.f;
    return w * vl[((size_t)y * WBEV + x) * CDIM];
}

__global__ __launch_bounds__(256)
void msda_kernel(const float* __restrict__ val,    // [NV][256]
                 const float* __restrict__ off,    // [NQ][512]
                 const float* __restrict__ aw,     // [NQ][128]
                 const float* __restrict__ refp,   // [NQ][4][2]
                 float* __restrict__ out) {        // [NQ][256]
    int wave = threadIdx.x >> 5, lane = threadIdx.x & 31;
    int wid = blockIdx.x * 8 + wave;  // 0..NQ*HEADS-1
    int q = wid >> 3, h = wid & 7;

    // softmax over the 16 (level,point) attention weights for this (q,h)
    float a = aw[(size_t)q * 128 + h * 16 + (lane & 15)];
    float mx = a;
    #pragma unroll
    for (int m = 8; m >= 1; m >>= 1) mx = fmaxf(mx, __shfl_xor(mx, m, 32));
    float e = __expf(a - mx);
    float se = e;
    #pragma unroll
    for (int m = 8; m >= 1; m >>= 1) se += __shfl_xor(se, m, 32);
    float wl = e / se;

    const float* offq = off + (size_t)q * 512 + h * 64;  // [l][p][2]
    float acc = 0.f;
    #pragma unroll
    for (int lv = 0; lv < LVLS; ++lv) {
        float rx = refp[(size_t)q * 8 + lv * 2 + 0];
        float ry = refp[(size_t)q * 8 + lv * 2 + 1];
        const float* vl = val + (size_t)lv * (HBEV * WBEV) * CDIM + h * DH + lane;
        __builtin_prefetch(vl, 0, 1);   // global_prefetch_b8
        #pragma unroll
        for (int p = 0; p < PTS; ++p) {
            int lp = lv * 4 + p;
            float w_lp = __shfl(wl, lp, 32);
            float lx = rx + offq[lp * 2 + 0] * (1.f / (float)WBEV);
            float ly = ry + offq[lp * 2 + 1] * (1.f / (float)HBEV);
            // grid = loc*2-1; x = (g+1)*W/2 - 0.5 = loc*W - 0.5
            float x = lx * (float)WBEV - 0.5f;
            float y = ly * (float)HBEV - 0.5f;
            float x0f = floorf(x), y0f = floorf(y);
            int x0 = (int)x0f, y0 = (int)y0f;
            float wx1 = x - x0f, wy1 = y - y0f;
            float wx0 = 1.f - wx1, wy0 = 1.f - wy1;
            float s = 0.f;
            s += msda_tap(vl, x0,     y0,     wx0 * wy0);
            s += msda_tap(vl, x0 + 1, y0,     wx1 * wy0);
            s += msda_tap(vl, x0,     y0 + 1, wx0 * wy1);
            s += msda_tap(vl, x0 + 1, y0 + 1, wx1 * wy1);
            acc += w_lp * s;
        }
    }
    out[(size_t)q * CDIM + h * DH + lane] = acc;
}

// ---------------------------------------------------------------------------
// Host-side orchestration
// ---------------------------------------------------------------------------
extern "C" void kernel_launch(void* const* d_in, const int* in_sizes, int n_in,
                              void* d_out, int out_size, void* d_ws, size_t ws_size,
                              hipStream_t stream) {
    (void)in_sizes; (void)n_in; (void)out_size; (void)ws_size;
    const float* query  = (const float*)d_in[0];
    const float* hist   = (const float*)d_in[1];
    const float* refp   = (const float*)d_in[2];
    const float* ln1_g  = (const float*)d_in[5];
    const float* ln1_b  = (const float*)d_in[6];
    const float* w_qkv  = (const float*)d_in[7];
    const float* w_sa   = (const float*)d_in[8];
    const float* b_sa   = (const float*)d_in[9];
    const float* ln2_g  = (const float*)d_in[10];
    const float* ln2_b  = (const float*)d_in[11];
    const float* w_off  = (const float*)d_in[12];
    const float* b_off  = (const float*)d_in[13];
    const float* w_aw   = (const float*)d_in[14];
    const float* b_aw   = (const float*)d_in[15];
    const float* w_val  = (const float*)d_in[16];
    const float* b_val  = (const float*)d_in[17];
    const float* w_cout = (const float*)d_in[18];
    const float* b_cout = (const float*)d_in[19];
    const float* ln3_g  = (const float*)d_in[20];
    const float* ln3_b  = (const float*)d_in[21];
    const float* w_ff1  = (const float*)d_in[22];
    const float* b_ff1  = (const float*)d_in[23];
    const float* w_ff2  = (const float*)d_in[24];
    const float* b_ff2  = (const float*)d_in[25];
    float* out = (float*)d_out;

    char* base = (char*)d_ws;
    size_t ofs = 0;
    auto alloc = [&](size_t bytes) -> void* {
        void* p = base + ofs;
        ofs = (ofs + bytes + 255) & ~(size_t)255;
        return p;
    };
    typedef unsigned short bf;
    bf* wqkvT = (bf*)alloc((size_t)768 * 256 * 2);
    bf* wsaT  = (bf*)alloc((size_t)256 * 256 * 2);
    bf* woffT = (bf*)alloc((size_t)512 * 256 * 2);
    bf* wawT  = (bf*)alloc((size_t)128 * 256 * 2);
    bf* wvalT = (bf*)alloc((size_t)256 * 256 * 2);
    bf* wcoT  = (bf*)alloc((size_t)256 * 256 * 2);
    bf* wff1T = (bf*)alloc((size_t)1024 * 256 * 2);
    bf* wff2T = (bf*)alloc((size_t)256 * 1024 * 2);
    bf*    xn    = (bf*)alloc((size_t)NQ * CDIM * 2);
    float* qkv   = (float*)alloc((size_t)NQ * 768 * 4);
    bf*    Qh    = (bf*)alloc((size_t)HEADS * NQ * DH * 2);
    bf*    Kh    = (bf*)alloc((size_t)HEADS * NQ * DH * 2);
    bf*    Vt    = (bf*)alloc((size_t)HEADS * DH * NQ * 2);
    float* sa    = (float*)alloc((size_t)NQ * CDIM * 4);
    bf*    sa_b  = (bf*)alloc((size_t)NQ * CDIM * 2);
    float* x1    = (float*)alloc((size_t)NQ * CDIM * 4);
    bf*    qn    = (bf*)alloc((size_t)NQ * CDIM * 2);
    bf*    vn    = (bf*)alloc((size_t)NV * CDIM * 2);
    float* val   = (float*)alloc((size_t)NV * CDIM * 4);
    float* offb  = (float*)alloc((size_t)NQ * 512 * 4);
    float* awb   = (float*)alloc((size_t)NQ * 128 * 4);
    float* cross = (float*)alloc((size_t)NQ * CDIM * 4);
    bf*    cr_b  = (bf*)alloc((size_t)NQ * CDIM * 2);
    float* x2    = (float*)alloc((size_t)NQ * CDIM * 4);
    bf*    xn3   = (bf*)alloc((size_t)NQ * CDIM * 2);
    float* ff1   = (float*)alloc((size_t)NQ * FFD * 4);
    bf*    ff1_b = (bf*)alloc((size_t)NQ * FFD * 2);

    // --- weight transpose+convert ---
    wtr_kernel<<<(256 * 768 + 255) / 256, 256, 0, stream>>>(w_qkv, wqkvT, 256, 768);
    wtr_kernel<<<(256 * 256 + 255) / 256, 256, 0, stream>>>(w_sa, wsaT, 256, 256);
    wtr_kernel<<<(256 * 512 + 255) / 256, 256, 0, stream>>>(w_off, woffT, 256, 512);
    wtr_kernel<<<(256 * 128 + 255) / 256, 256, 0, stream>>>(w_aw, wawT, 256, 128);
    wtr_kernel<<<(256 * 256 + 255) / 256, 256, 0, stream>>>(w_val, wvalT, 256, 256);
    wtr_kernel<<<(256 * 256 + 255) / 256, 256, 0, stream>>>(w_cout, wcoT, 256, 256);
    wtr_kernel<<<(256 * 1024 + 255) / 256, 256, 0, stream>>>(w_ff1, wff1T, 256, 1024);
    wtr_kernel<<<(1024 * 256 + 255) / 256, 256, 0, stream>>>(w_ff2, wff2T, 1024, 256);

    // --- self-attention (PreNorm) ---
    ln_bf16_kernel<<<NQ / 8, 256, 0, stream>>>(query, ln1_g, ln1_b, xn, NQ);
    gemm_bf16_kernel<<<dim3(NQ / 128, 768 / 64), 256, 0, stream>>>(
        xn, wqkvT, nullptr, nullptr, qkv, NQ, 768, 256);
    qkv_split_kernel<<<NQ * CDIM / 256, 256, 0, stream>>>(qkv, Qh, Kh, Vt);
    attn_kernel<<<dim3(NQ / 128, HEADS), 256, 0, stream>>>(Qh, Kh, Vt, sa);
    cvt_kernel<<<NQ * CDIM / 256, 256, 0, stream>>>(sa, sa_b, NQ * CDIM);
    gemm_bf16_kernel<<<dim3(NQ / 128, 256 / 64), 256, 0, stream>>>(
        sa_b, wsaT, b_sa, query, x1, NQ, 256, 256);   // x1 = sa@Wo + b + query

    // --- deformable cross-attention (shared LN2 on q and value) ---
    ln_bf16_kernel<<<NQ / 8, 256, 0, stream>>>(x1, ln2_g, ln2_b, qn, NQ);
    ln_bf16_kernel<<<NV / 8, 256, 0, stream>>>(hist, ln2_g, ln2_b, vn, NV);
    gemm_bf16_kernel<<<dim3(NV / 128, 256 / 64), 256, 0, stream>>>(
        vn, wvalT, b_val, nullptr, val, NV, 256, 256);
    gemm_bf16_kernel<<<dim3(NQ / 128, 512 / 64), 256, 0, stream>>>(
        qn, woffT, b_off, nullptr, offb, NQ, 512, 256);
    gemm_bf16_kernel<<<dim3(NQ / 128, 128 / 64), 256, 0, stream>>>(
        qn, wawT, b_aw, nullptr, awb, NQ, 128, 256);
    msda_kernel<<<NQ * HEADS / 8, 256, 0, stream>>>(val, offb, awb, refp, cross);
    cvt_kernel<<<NQ * CDIM / 256, 256, 0, stream>>>(cross, cr_b, NQ * CDIM);
    gemm_bf16_kernel<<<dim3(NQ / 128, 256 / 64), 256, 0, stream>>>(
        cr_b, wcoT, b_cout, x1, x2, NQ, 256, 256);    // x2 = cross@Wc + b + x1

    // --- feed-forward (PreNorm, no nonlinearity) ---
    ln_bf16_kernel<<<NQ / 8, 256, 0, stream>>>(x2, ln3_g, ln3_b, xn3, NQ);
    gemm_bf16_kernel<<<dim3(NQ / 128, FFD / 64), 256, 0, stream>>>(
        xn3, wff1T, b_ff1, nullptr, ff1, NQ, FFD, 256);
    cvt_kernel<<<NQ * FFD / 256, 256, 0, stream>>>(ff1, ff1_b, NQ * FFD);
    gemm_bf16_kernel<<<dim3(NQ / 128, 256 / 64), 256, 0, stream>>>(
        ff1_b, wff2T, b_ff2, x2, out, NQ, 256, FFD);  // out = ff + x2
}
